// MultiheadAttention_49366354100565
// MI455X (gfx1250) — matrix operand order repaired and checked by reading, not compiled, not run on hardware
//
#include <hip/hip_runtime.h>

// ---------------------------------------------------------------------------
// Problem constants (from reference): B=2, T=2048, E=1024, H=16, D=64
// ---------------------------------------------------------------------------
#define BATCH 2
#define SEQ   2048
#define EMB   1024
#define HEADS 16
#define HDIM  64
#define BH    (BATCH * HEADS)   // 32
#define BT    (BATCH * SEQ)     // 4096

typedef __attribute__((ext_vector_type(16))) _Float16 v16h;
typedef __attribute__((ext_vector_type(8)))  _Float16 v8h;
typedef __attribute__((ext_vector_type(4)))  _Float16 v4h;
typedef __attribute__((ext_vector_type(8)))  float    v8f;
typedef unsigned int u32x4 __attribute__((ext_vector_type(4)));
typedef int          i32x8 __attribute__((ext_vector_type(8)));
typedef int          i32x4 __attribute__((ext_vector_type(4)));

// D = A(16x32 f16) * B(32x16 f16) + C(16x16 f32)  -> v_wmma_f32_16x16x32_f16
__device__ __forceinline__ v8f wmma16(v16h a, v16h b, v8f c) {
    return __builtin_amdgcn_wmma_f32_16x16x32_f16(
        /*neg_a=*/false, a, /*neg_b=*/false, b,
        /*c_mod=*/(short)0, c, /*reuse_a=*/false, /*reuse_b=*/false);
}

// Assemble a 16-half fragment from two 16-byte chunks (ds_load_b128 /
// global_load_b128 each).
__device__ __forceinline__ v16h load_frag(const _Float16* p0, const _Float16* p1) {
    v8h lo = *(const v8h*)p0;
    v8h hi = *(const v8h*)p1;
    v16h r;
#pragma unroll
    for (int i = 0; i < 8; ++i) { r[i] = lo[i]; r[i + 8] = hi[i]; }
    return r;
}

// ---------------------------------------------------------------------------
// Tensor Data Mover: 2-D tile load global->LDS (ISA ch.8 / D# groups 0..3).
// rows x cols elements of 2 bytes, row-major, contiguous into LDS.
// Issue from ONE wave only; follow with s_wait_tensorcnt(0).
// Toolchain uses the 6-arg builtin: (g0, g1, g2, g3, g_extra, cpol).
// ---------------------------------------------------------------------------
__device__ __forceinline__ void tdm_load_2d_f16(unsigned lds_byte_addr,
                                                const void* gptr,
                                                unsigned cols, unsigned rows,
                                                unsigned row_stride_elems) {
    const unsigned long long ga = (unsigned long long)(uintptr_t)gptr;
    u32x4 g0;
    g0[0] = 1u;                                   // count=1, user descriptor
    g0[1] = lds_byte_addr;                        // lds_addr [63:32]
    g0[2] = (unsigned)(ga & 0xffffffffu);         // global_addr [95:64]
    g0[3] = (unsigned)((ga >> 32) & 0x01ffffffu)  // global_addr [120:96]
            | (2u << 30);                         // type=2 ("image") [127:126]
    i32x8 g1;
    g1[0] = (int)(1u << 16);                      // wg_mask=0, data_size=1 (2B)
    g1[1] = (int)(cols << 16);                    // tensor_dim0[15:0]  @[63:48]
    g1[2] = (int)(rows << 16);                    // td0 hi=0; tensor_dim1[15:0]
    g1[3] = (int)(cols << 16);                    // td1 hi=0; tile_dim0
    g1[4] = (int)rows;                            // tile_dim1; tile_dim2=0
    g1[5] = (int)row_stride_elems;                // tensor_dim0_stride[31:0]
    g1[6] = 0;                                    // stride0 hi, stride1 lo
    g1[7] = 0;                                    // stride1 hi
    const i32x4 z4 = {0, 0, 0, 0};                // groups 2/3 unused (2-D)
    const i32x8 z8 = {0, 0, 0, 0, 0, 0, 0, 0};
    __builtin_amdgcn_tensor_load_to_lds(g0, g1, z4, z4, z8, 0);
}

// ---------------------------------------------------------------------------
// Generic C[M,N] = (A[M,K] @ W[N,K]^T + bias) * alpha     (f32 in, f32 out)
// A and W converted f32->f16 on the global->LDS staging path; compute on
// v_wmma_f32_16x16x32_f16. Tile 128x128, 8 waves (256 thr), wave = 32x64.
// M,N,K multiples of 128 here (4096/1024/1024).
// ---------------------------------------------------------------------------
__global__ __launch_bounds__(256)
void gemm_xw_kernel(const float* __restrict__ A, const float* __restrict__ W,
                    const float* __restrict__ bias, float* __restrict__ C,
                    int M, int N, int K, float alpha) {
    __shared__ _Float16 Alds[128 * 32];
    __shared__ _Float16 Blds[128 * 32];

    const int tid   = threadIdx.x;
    const int lane  = tid & 31;
    const int wave  = tid >> 5;
    const int mw    = (wave & 3) * 32;   // wave row offset in tile
    const int nw    = (wave >> 2) * 64;  // wave col offset in tile
    const int Mbase = blockIdx.y * 128;
    const int Nbase = blockIdx.x * 128;

    const int lrow = tid >> 3;        // 0..31
    const int lcol = (tid & 7) * 4;   // 0..28 step 4

    v8f acc[2][4] = {};

    for (int k0 = 0; k0 < K; k0 += 32) {
#pragma unroll
        for (int i = 0; i < 4; ++i) {
            const int r = lrow + i * 32;
            const float* ap = A + (size_t)(Mbase + r) * K + k0 + lcol;
            const float* wp = W + (size_t)(Nbase + r) * K + k0 + lcol;
            const float4 av = *(const float4*)ap;
            v4h ah = { (_Float16)av.x, (_Float16)av.y, (_Float16)av.z, (_Float16)av.w };
            *(v4h*)(Alds + r * 32 + lcol) = ah;
            const float4 wv = *(const float4*)wp;
            v4h wh = { (_Float16)wv.x, (_Float16)wv.y, (_Float16)wv.z, (_Float16)wv.w };
            *(v4h*)(Blds + r * 32 + lcol) = wh;
            // speculative prefetch of next K-tile (safe past end; dropped)
            __builtin_prefetch(ap + 32, 0, 1);
            __builtin_prefetch(wp + 32, 0, 1);
        }
        __syncthreads();

        const int row = lane & 15;
        const int hh  = lane >> 4;
        v16h af[2], bf[4];
#pragma unroll
        for (int mt = 0; mt < 2; ++mt) {
            const _Float16* p = Alds + (mw + mt * 16 + row) * 32 + hh * 8;
            af[mt] = load_frag(p, p + 16);       // K = {h*8..+7, h*8+16..+23}
        }
#pragma unroll
        for (int nt = 0; nt < 4; ++nt) {
            const _Float16* p = Blds + (nw + nt * 16 + row) * 32 + hh * 8;
            bf[nt] = load_frag(p, p + 16);       // K = {h*8..+7, h*8+16..+23}
        }
#pragma unroll
        for (int mt = 0; mt < 2; ++mt)
#pragma unroll
            for (int nt = 0; nt < 4; ++nt)
                acc[mt][nt] = wmma16(af[mt], bf[nt], acc[mt][nt]);
        __syncthreads();
    }

    // Epilogue: C-layout (ISA 7.12.2): col = lane&15 ; VGPR r -> row r (+8 if lane>=16)
    const int ncol = lane & 15;
    const int hi8  = (lane >> 4) * 8;
#pragma unroll
    for (int mt = 0; mt < 2; ++mt)
#pragma unroll
        for (int nt = 0; nt < 4; ++nt) {
            const int n = Nbase + nw + nt * 16 + ncol;
            const float bn = bias[n];
#pragma unroll
            for (int r = 0; r < 8; ++r) {
                const int m = Mbase + mw + mt * 16 + r + hi8;
                C[(size_t)m * N + n] = (acc[mt][nt][r] + bn) * alpha;
            }
        }
}

// ---------------------------------------------------------------------------
// Exact top-512-of-1024 row pruning via 4-bit radix select on monotone keys.
// One block of 1024 threads per row; LDS histogram atomics.
// ---------------------------------------------------------------------------
__global__ __launch_bounds__(1024)
void prune_topk_kernel(float* __restrict__ x) {
    const int row = blockIdx.x;
    const int tid = threadIdx.x;
    const float v = x[(size_t)row * EMB + tid];
    unsigned u = __float_as_uint(v);
    u = (u & 0x80000000u) ? ~u : (u | 0x80000000u);  // order-preserving key

    __shared__ unsigned hist[16];
    __shared__ unsigned prefix_s;
    __shared__ unsigned kneed_s;
    if (tid == 0) { prefix_s = 0u; kneed_s = 512u; }
    __syncthreads();

    for (int shift = 28; shift >= 0; shift -= 4) {
        if (tid < 16) hist[tid] = 0u;
        __syncthreads();
        const bool match = (shift == 28) ||
                           ((u >> (shift + 4)) == (prefix_s >> (shift + 4)));
        if (match) atomicAdd(&hist[(u >> shift) & 15u], 1u);
        __syncthreads();
        if (tid == 0) {
            unsigned kn = kneed_s;
            int d = 15;
            for (; d > 0; --d) {
                const unsigned c = hist[d];
                if (kn <= c) break;
                kn -= c;
            }
            kneed_s  = kn;
            prefix_s |= ((unsigned)d) << shift;
        }
        __syncthreads();
    }
    const bool keep = (u >= prefix_s);  // ties at threshold all kept
    x[(size_t)row * EMB + tid] = keep ? v : 0.0f;
}

// ---------------------------------------------------------------------------
// xPos rotary + head split + f32->f16. q16/k16 stored [bh][t][64];
// v stored transposed [bh][64][t] so P@v's B operand has contiguous K rows.
// grid = (T, BH), block = 64.
// ---------------------------------------------------------------------------
__global__ __launch_bounds__(64)
void xpos_split_kernel(const float* __restrict__ q32, const float* __restrict__ k32,
                       const float* __restrict__ v32,
                       _Float16* __restrict__ q16, _Float16* __restrict__ k16,
                       _Float16* __restrict__ vT16) {
    const int t  = blockIdx.x;
    const int bh = blockIdx.y;
    const int b  = bh >> 4;
    const int h  = bh & 15;
    const int d  = threadIdx.x;
    const size_t src = ((size_t)b * SEQ + t) * EMB + h * HDIM;

    vT16[((size_t)bh * HDIM + d) * SEQ + t] = (_Float16)v32[src + d];

    if (d < 32) {
        const int j = d;
        const float base_scale = (2.0f * j + 0.4f * HDIM) / (1.4f * HDIM);
        const float power      = ((float)t - (float)(SEQ / 2)) * (1.0f / 512.0f);
        const float sc_q = powf(base_scale, power);
        const float sc_k = 1.0f / sc_q;                       // downscale=True
        const float inv_freq = powf(10000.0f, -(float)j * (1.0f / 32.0f));
        const float ang = (float)t * inv_freq;
        const float s = sinf(ang), c = cosf(ang);

        const size_t dst = ((size_t)bh * SEQ + t) * HDIM + 2 * j;
        {   // q (already scaled by D^-1/2 in projection)
            const float x0 = q32[src + 2 * j], x1 = q32[src + 2 * j + 1];
            const float cs = c * sc_q, ss = s * sc_q;
            q16[dst]     = (_Float16)(x0 * cs - x1 * ss);
            q16[dst + 1] = (_Float16)(x1 * cs + x0 * ss);
        }
        {   // k
            const float x0 = k32[src + 2 * j], x1 = k32[src + 2 * j + 1];
            const float cs = c * sc_k, ss = s * sc_k;
            k16[dst]     = (_Float16)(x0 * cs - x1 * ss);
            k16[dst + 1] = (_Float16)(x1 * cs + x0 * ss);
        }
    }
}

// ---------------------------------------------------------------------------
// Fused attention: per WG -> one head (bh) x 16 query rows.
//   q tile staged via Tensor Data Mover (tensor_load_to_lds + s_wait_tensorcnt)
//   S(16x2048) via WMMA -> LDS (f32, 128 KB; only possible w/ 320KB WGP LDS)
//   softmax in LDS, w16 written once to d_out (the 268MB bandwidth floor)
//   attn(16x64) = P(f16 LDS) @ vT via WMMA, cross-wave reduce in LDS.
// 4 waves; wave w owns s-slice [w*512, w*512+512).
// ---------------------------------------------------------------------------
#define ATTN_SMEM_BYTES (16*SEQ*4 + 16*SEQ*2 + 16*HDIM*2 + 4*16*HDIM*4)  // 215040

__global__ __launch_bounds__(128)
void attention_kernel(const _Float16* __restrict__ q16,
                      const _Float16* __restrict__ k16,
                      const _Float16* __restrict__ vT16,
                      _Float16* __restrict__ wOut,
                      float* __restrict__ attn32) {
    extern __shared__ __align__(16) char smem[];
    float*    S    = (float*)smem;                          // 16*SEQ f32
    _Float16* P    = (_Float16*)(smem + 16 * SEQ * 4);      // 16*SEQ f16
    _Float16* Qs   = P + 16 * SEQ;                          // 16*64 f16
    float*    Cred = (float*)(Qs + 16 * HDIM);              // 4*16*64 f32

    const int tid  = threadIdx.x;
    const int lane = tid & 31;
    const int w    = tid >> 5;          // wave id 0..3
    const int rb   = blockIdx.x;        // row block (T/16)
    const int bh   = blockIdx.y;
    const int b    = bh >> 4;
    const int h    = bh & 15;

    // stage 16x64 q rows into LDS via TDM (wave 0 issues; TENSORcnt is
    // per-wave so wave 0 waits, then the WG barrier publishes the LDS data).
    if (w == 0) {
        const _Float16* qg = q16 + ((size_t)bh * SEQ + rb * 16) * HDIM;
        // flat LDS aperture: low 32 bits of the generic pointer = LDS offset
        tdm_load_2d_f16((unsigned)(uintptr_t)Qs, qg,
                        /*cols=*/HDIM, /*rows=*/16, /*row_stride=*/HDIM);
        __builtin_amdgcn_s_wait_tensorcnt(0);
    }
    __syncthreads();

    const int s_begin = w * (SEQ / 4);

    // ---- scores: S[0:16, slice] = q @ k^T --------------------------------
    for (int st = s_begin; st < s_begin + SEQ / 4; st += 16) {
        v8f acc = {};
        const int row = lane & 15;
        const int hh  = lane >> 4;
#pragma unroll
        for (int kk = 0; kk < HDIM; kk += 32) {
            const _Float16* ap = Qs + row * HDIM + kk + hh * 8;
            v16h a = load_frag(ap, ap + 16);
            const _Float16* bp =
                k16 + ((size_t)bh * SEQ + st + row) * HDIM + kk + hh * 8;
            v16h bf = load_frag(bp, bp + 16);
            acc = wmma16(a, bf, acc);
        }
        const int hi8 = hh * 8;
        const int col = st + row;
#pragma unroll
        for (int r = 0; r < 8; ++r) S[(r + hi8) * SEQ + col] = acc[r];
    }
    __syncthreads();

    // ---- softmax (f32) + fp16 quantize; wave w owns rows 4w..4w+3 --------
    for (int rr = 0; rr < 4; ++rr) {
        const int row = w * 4 + rr;
        float mx = -3.0e38f;
        for (int c = lane; c < SEQ; c += 32) mx = fmaxf(mx, S[row * SEQ + c]);
#pragma unroll
        for (int off = 16; off; off >>= 1) mx = fmaxf(mx, __shfl_xor(mx, off, 32));
        float sum = 0.0f;
        for (int c = lane; c < SEQ; c += 32) sum += __expf(S[row * SEQ + c] - mx);
#pragma unroll
        for (int off = 16; off; off >>= 1) sum += __shfl_xor(sum, off, 32);
        const float inv = 1.0f / sum;
        // attn_weights laid out (H, B, T, T)
        const size_t wbase = (((size_t)(h * BATCH + b) * SEQ) + (rb * 16 + row)) * SEQ;
        for (int c = lane; c < SEQ; c += 32) {
            const _Float16 p = (_Float16)(__expf(S[row * SEQ + c] - mx) * inv);
            P[row * SEQ + c] = p;
            wOut[wbase + c]  = p;
        }
    }
    __syncthreads();

    // ---- attn = P @ v : wave w reduces its K-slice; 4 n-tiles of 16 ------
    v8f acc2[4] = {};
    for (int s0 = s_begin; s0 < s_begin + SEQ / 4; s0 += 32) {
        const int row = lane & 15;
        const int hh  = lane >> 4;
        const _Float16* ap = P + row * SEQ + s0 + hh * 8;
        v16h a = load_frag(ap, ap + 16);
#pragma unroll
        for (int nt = 0; nt < 4; ++nt) {
            const _Float16* bp =
                vT16 + ((size_t)bh * HDIM + nt * 16 + row) * SEQ + s0 + hh * 8;
            v16h bf = load_frag(bp, bp + 16);
            acc2[nt] = wmma16(a, bf, acc2[nt]);
        }
    }
    {
        const int ncol = lane & 15;
        const int hi8  = (lane >> 4) * 8;
#pragma unroll
        for (int nt = 0; nt < 4; ++nt)
#pragma unroll
            for (int r = 0; r < 8; ++r)
                Cred[(w * 16 + r + hi8) * HDIM + nt * 16 + ncol] = acc2[nt][r];
    }
    __syncthreads();

    // cross-wave reduce + head-merge write: attn32 is [B, T, E]
    for (int i = tid; i < 16 * HDIM; i += 128) {
        const int m = i >> 6, n = i & 63;
        const float s = Cred[m * HDIM + n] + Cred[(16 + m) * HDIM + n] +
                        Cred[(32 + m) * HDIM + n] + Cred[(48 + m) * HDIM + n];
        attn32[(((size_t)b * SEQ) + rb * 16 + m) * EMB + h * HDIM + n] = s;
    }
}

// ---------------------------------------------------------------------------
// Host launcher
// ---------------------------------------------------------------------------
extern "C" void kernel_launch(void* const* d_in, const int* in_sizes, int n_in,
                              void* d_out, int out_size, void* d_ws, size_t ws_size,
                              hipStream_t stream) {
    (void)in_sizes; (void)n_in; (void)out_size; (void)ws_size;

    const float* query = (const float*)d_in[0];
    const float* key   = (const float*)d_in[1];
    const float* value = (const float*)d_in[2];
    const float* Wq    = (const float*)d_in[3];
    const float* bq    = (const float*)d_in[4];
    const float* Wk    = (const float*)d_in[5];
    const float* bk    = (const float*)d_in[6];
    const float* Wv    = (const float*)d_in[7];
    const float* bv    = (const float*)d_in[8];
    const float* Wo    = (const float*)d_in[9];
    const float* bo    = (const float*)d_in[10];

    // d_out = [out f32 (B,T,E)] ++ [attn_weights fp16 (H,B,T,T)]
    float*    outF = (float*)d_out;
    _Float16* wOut = (_Float16*)((char*)d_out + (size_t)BT * EMB * sizeof(float));

    // workspace carve-up
    char* ws = (char*)d_ws;
    const size_t F32MAT = (size_t)BT * EMB * sizeof(float);     // 16 MB
    const size_t F16HED = (size_t)BH * SEQ * HDIM * 2;          // 8 MB
    float*    q32    = (float*)(ws + 0 * F32MAT);
    float*    k32    = (float*)(ws + 1 * F32MAT);
    float*    v32    = (float*)(ws + 2 * F32MAT);
    float*    attn32 = (float*)(ws + 3 * F32MAT);
    _Float16* q16    = (_Float16*)(ws + 4 * F32MAT);
    _Float16* k16    = (_Float16*)(ws + 4 * F32MAT + 1 * F16HED);
    _Float16* vT16   = (_Float16*)(ws + 4 * F32MAT + 2 * F16HED);

    const dim3 ggrid(EMB / 128, BT / 128);  // (8, 32)

    // projections (q folds in D^-1/2 = 0.125; uniform positive scale commutes
    // with the top-k mask)
    gemm_xw_kernel<<<ggrid, 256, 0, stream>>>(query, Wq, bq, q32, BT, EMB, EMB, 0.125f);
    gemm_xw_kernel<<<ggrid, 256, 0, stream>>>(key,   Wk, bk, k32, BT, EMB, EMB, 1.0f);
    gemm_xw_kernel<<<ggrid, 256, 0, stream>>>(value, Wv, bv, v32, BT, EMB, EMB, 1.0f);

    // exact per-row top-512 pruning
    prune_topk_kernel<<<BT, 1024, 0, stream>>>(q32);
    prune_topk_kernel<<<BT, 1024, 0, stream>>>(k32);
    prune_topk_kernel<<<BT, 1024, 0, stream>>>(v32);

    // xPos + head split + f16 (v transposed)
    xpos_split_kernel<<<dim3(SEQ, BH), 64, 0, stream>>>(q32, k32, v32, q16, k16, vT16);

    // fused scores/softmax/PV (215 KB dynamic LDS per WG)
    (void)hipFuncSetAttribute(reinterpret_cast<const void*>(attention_kernel),
                              hipFuncAttributeMaxDynamicSharedMemorySize,
                              ATTN_SMEM_BYTES);
    attention_kernel<<<dim3(SEQ / 16, BH), 128, ATTN_SMEM_BYTES, stream>>>(
        q16, k16, vT16, wOut, attn32);

    // output projection
    gemm_xw_kernel<<<ggrid, 256, 0, stream>>>(attn32, Wo, bo, outF, BT, EMB, EMB, 1.0f);
}
